// CNAttnProcessor_57939108823381
// MI455X (gfx1250) — matrix operand order repaired and checked by reading, not compiled, not run on hardware
//
#include <hip/hip_runtime.h>
#include <hip/hip_bf16.h>

typedef __attribute__((ext_vector_type(16))) __bf16 v16bf;
typedef __attribute__((ext_vector_type(8)))  __bf16 v8bf;
typedef __attribute__((ext_vector_type(2)))  __bf16 v2bf;
typedef __attribute__((ext_vector_type(8)))  float  v8f;
typedef __attribute__((ext_vector_type(4)))  float  v4f;
typedef __attribute__((ext_vector_type(2)))  float  v2f;
typedef __attribute__((ext_vector_type(4)))  unsigned int u32x4;
typedef __attribute__((ext_vector_type(8)))  int  i32x8;
typedef __attribute__((ext_vector_type(4)))  int  i32x4;

#define WMMA_BF16(a, b, c) \
  __builtin_amdgcn_wmma_f32_16x16x32_bf16(false, (a), false, (b), (short)0, (c), false, false)

// Convert two floats -> packed bf16 pair, single ds_store_b32 (p must be 4B aligned).
__device__ __forceinline__ void st2bf(__bf16* p, float a, float b) {
  v2bf t;
  t[0] = (__bf16)a;
  t[1] = (__bf16)b;
  *(v2bf*)p = t;
}

// A-matrix (16x32 bf16) fragment from LDS tile stored row-major [16][stride].
// ISA layout: lane (m = l&15, hi = l>>4) holds K = {hi*8+0..7, 16+hi*8+0..7}.
__device__ __forceinline__ v16bf ldsLoadA(const __bf16* p, int stride) {
  const int lane = threadIdx.x & 31;
  const int m = lane & 15, hi = lane >> 4;
  const __bf16* r = p + m * stride;
  v16bf a;
#pragma unroll
  for (int j = 0; j < 8; ++j) a[j] = r[hi * 8 + j];
#pragma unroll
  for (int j = 0; j < 8; ++j) a[8 + j] = r[16 + hi * 8 + j];
  return a;
}

// B-matrix (32x16 bf16) fragment from LDS tile stored column-major [16 n][stride k].
// ISA layout: lane (n = l&15, hi = l>>4) holds K = {hi*8+0..7, 16+hi*8+0..7} of column n.
__device__ __forceinline__ v16bf ldsLoadB(const __bf16* p, int stride) {
  const int lane = threadIdx.x & 31;
  const int n = lane & 15, hi = lane >> 4;
  const __bf16* r = p + n * stride;
  v16bf b;
#pragma unroll
  for (int j = 0; j < 8; ++j) b[j] = r[hi * 8 + j];
#pragma unroll
  for (int j = 0; j < 8; ++j) b[8 + j] = r[16 + hi * 8 + j];
  return b;
}

// ---------------- Kernel 1: Q projection ----------------
// q[b, s, n] = sum_c x[b, c, s] * Wq[c, n]   (x = hidden_states, channel-major)
__global__ __launch_bounds__(256) void k_qproj(const float* __restrict__ hs,
                                               const float* __restrict__ Wq,
                                               __bf16* __restrict__ qbuf) {
  __shared__ __bf16 As[128 * 32];
  __shared__ __bf16 Bs[64 * 32];
  const int s0 = blockIdx.x * 128;
  const int n0 = blockIdx.y * 64;
  const int b  = blockIdx.z;
  const int tid = threadIdx.x;
  const int wave = tid >> 5, lane = tid & 31;
  const int nl = lane & 15, hi = lane >> 4;

  v8f acc[4] = {};
  for (int kt = 0; kt < 10; ++kt) {
    const int k0 = kt * 32;
    if (kt + 1 < 10)  // CDNA5 global_prefetch_b8 of next A tile
      __builtin_prefetch(hs + (size_t)(b * 320 + k0 + 32 + (tid >> 3)) * 4096 + s0 + (tid & 7) * 16, 0, 1);
#pragma unroll
    for (int it = 0; it < 8; ++it) {            // A: 128 si x 16 kk-pairs (packed b32 stores)
      int idx = tid + it * 256;
      int si = idx & 127, kk = (idx >> 7) * 2;
      float f0 = hs[(size_t)(b * 320 + k0 + kk)     * 4096 + s0 + si];
      float f1 = hs[(size_t)(b * 320 + k0 + kk + 1) * 4096 + s0 + si];
      st2bf(As + si * 32 + kk, f0, f1);
    }
#pragma unroll
    for (int it = 0; it < 4; ++it) {            // B: 64 n x 16 kk-pairs
      int idx = tid + it * 256;
      int n = idx & 63, kk = (idx >> 6) * 2;
      st2bf(Bs + n * 32 + kk, Wq[(k0 + kk) * 320 + n0 + n], Wq[(k0 + kk + 1) * 320 + n0 + n]);
    }
    __syncthreads();
    v16bf a = ldsLoadA(As + wave * 16 * 32, 32);
#pragma unroll
    for (int nt = 0; nt < 4; ++nt) {
      v16bf bb = ldsLoadB(Bs + nt * 16 * 32, 32);
      acc[nt] = WMMA_BF16(a, bb, acc[nt]);
    }
    __syncthreads();
  }
#pragma unroll
  for (int nt = 0; nt < 4; ++nt)
#pragma unroll
    for (int r = 0; r < 8; ++r) {
      int s = s0 + wave * 16 + r + 8 * hi;
      int c = n0 + nt * 16 + nl;
      qbuf[(size_t)(b * 4096 + s) * 320 + c] = (__bf16)acc[nt][r];
    }
}

// ---------------- Kernel 2: K/V projection ----------------
__global__ __launch_bounds__(160) void k_kvproj(const float* __restrict__ enc,
                                                const float* __restrict__ Wk,
                                                const float* __restrict__ Wv,
                                                __bf16* __restrict__ kbuf,
                                                __bf16* __restrict__ vbuf) {
  __shared__ __bf16 As[80 * 32];
  __shared__ __bf16 Bs[64 * 32];
  const int n0 = blockIdx.x * 64;
  const int sel = blockIdx.y;
  const int b  = blockIdx.z;
  const float* W = sel ? Wv : Wk;
  __bf16* out = sel ? vbuf : kbuf;
  const int tid = threadIdx.x;
  const int wave = tid >> 5, lane = tid & 31;
  const int nl = lane & 15, hi = lane >> 4;

  v8f acc[4] = {};
  for (int kt = 0; kt < 24; ++kt) {
    const int k0 = kt * 32;
    for (int idx = tid; idx < 80 * 16; idx += 160) {   // A: enc rows contiguous -> v2f loads
      int si = idx % 80, kk = (idx / 80) * 2;
      v2f f = {};
      if (si < 77) f = *(const v2f*)(enc + (size_t)(b * 93 + si) * 768 + k0 + kk);
      st2bf(As + si * 32 + kk, f[0], f[1]);
    }
    for (int idx = tid; idx < 64 * 16; idx += 160) {
      int n = idx & 63, kk = (idx >> 6) * 2;
      st2bf(Bs + n * 32 + kk, W[(k0 + kk) * 320 + n0 + n], W[(k0 + kk + 1) * 320 + n0 + n]);
    }
    __syncthreads();
    v16bf a = ldsLoadA(As + wave * 16 * 32, 32);
#pragma unroll
    for (int nt = 0; nt < 4; ++nt) {
      v16bf bb = ldsLoadB(Bs + nt * 16 * 32, 32);
      acc[nt] = WMMA_BF16(a, bb, acc[nt]);
    }
    __syncthreads();
  }
#pragma unroll
  for (int nt = 0; nt < 4; ++nt)
#pragma unroll
    for (int r = 0; r < 8; ++r) {
      int s = wave * 16 + r + 8 * hi;  // 0..79 (rows >= 77 are zero)
      out[(size_t)(b * 80 + s) * 320 + n0 + nt * 16 + nl] = (__bf16)acc[nt][r];
    }
}

// ---------------- Kernel 3: fused attention ----------------
__global__ __launch_bounds__(128) void k_attn(const __bf16* __restrict__ qbuf,
                                              const __bf16* __restrict__ kbuf,
                                              const __bf16* __restrict__ vbuf,
                                              __bf16* __restrict__ attnbuf) {
  __shared__ __bf16 Qs[64 * 64];     // [query][d: 40 padded to 64]
  __shared__ __bf16 Ks[80 * 64];     // [j][d]  (B tile for QK^T)
  __shared__ __bf16 Vt[48 * 96];     // [d pad 48][j pad 96] (B tile for PV)
  __shared__ __bf16 Ps[4][16 * 96];  // per-wave probs, A-layout staging
  const int s0 = blockIdx.x * 64;
  const int h  = blockIdx.y;
  const int b  = blockIdx.z;
  const int tid = threadIdx.x;
  const int wave = tid >> 5, lane = tid & 31;
  const int nl = lane & 15, hi = lane >> 4;
  const v8bf z8 = {};

  // Q/K staging: pure bf16 16-byte chunk copies (rows are 16B aligned: 40*2=80B)
#pragma unroll
  for (int it = 0; it < 4; ++it) {           // 64 si x 8 chunks
    int idx = tid + it * 128;
    int ci = idx & 7, si = idx >> 3;
    v8bf v = z8;
    if (ci < 5) v = *(const v8bf*)(qbuf + (size_t)(b * 4096 + s0 + si) * 320 + h * 40 + ci * 8);
    *(v8bf*)(Qs + si * 64 + ci * 8) = v;
  }
#pragma unroll
  for (int it = 0; it < 5; ++it) {           // 80 j x 8 chunks
    int idx = tid + it * 128;
    int ci = idx & 7, j = idx >> 3;
    v8bf v = z8;
    if (ci < 5) v = *(const v8bf*)(kbuf + (size_t)(b * 80 + j) * 320 + h * 40 + ci * 8);
    *(v8bf*)(Ks + j * 64 + ci * 8) = v;
  }
  for (int idx = tid; idx < 48 * 48; idx += 128) {  // Vt transpose: j-pairs packed
    int j = (idx % 48) * 2, d = idx / 48;
    v2bf t = {};
    if (d < 40) {
      if (j < 80)     t[0] = vbuf[(size_t)(b * 80 + j)     * 320 + h * 40 + d];
      if (j + 1 < 80) t[1] = vbuf[(size_t)(b * 80 + j + 1) * 320 + h * 40 + d];
    }
    *(v2bf*)(Vt + d * 96 + j) = t;
  }
  // zero probs K-pad (cols 80..95), 16B chunks
  {
    int idx = tid;  // 4 waves * 16 rows * 2 chunks = 128 == blockDim
    int w = idx >> 5, rem = idx & 31, m = rem >> 1, cp = rem & 1;
    *(v8bf*)(Ps[w] + m * 96 + 80 + cp * 8) = z8;
  }
  __syncthreads();

  // S = Q K^T : M=16/wave, N=80 (5 tiles), K=64 (2 chunks)
  v8f sacc[5] = {};
  v16bf a0 = ldsLoadA(Qs + wave * 16 * 64, 64);
  v16bf a1 = ldsLoadA(Qs + wave * 16 * 64 + 32, 64);
#pragma unroll
  for (int nt = 0; nt < 5; ++nt) {
    v16bf b0 = ldsLoadB(Ks + nt * 16 * 64, 64);
    v16bf b1 = ldsLoadB(Ks + nt * 16 * 64 + 32, 64);
    sacc[nt] = WMMA_BF16(a0, b0, sacc[nt]);
    sacc[nt] = WMMA_BF16(a1, b1, sacc[nt]);
  }
  const float scale = 0.15811388300841897f;  // 1/sqrt(40)
#pragma unroll
  for (int nt = 0; nt < 5; ++nt) {
    int j = nt * 16 + nl;
#pragma unroll
    for (int r = 0; r < 8; ++r) {
      float sv = sacc[nt][r] * scale;
      if (j >= 77) sv = -__builtin_inff();
      sacc[nt][r] = sv;
    }
  }
  // row softmax: row (r, hi) lives in lanes {hi*16 .. hi*16+15} -> shfl_xor width 16
#pragma unroll
  for (int r = 0; r < 8; ++r) {
    float mx = -__builtin_inff();
#pragma unroll
    for (int nt = 0; nt < 5; ++nt) mx = fmaxf(mx, sacc[nt][r]);
    for (int off = 1; off < 16; off <<= 1) mx = fmaxf(mx, __shfl_xor(mx, off, 16));
    float sum = 0.0f;
#pragma unroll
    for (int nt = 0; nt < 5; ++nt) {
      float p = __expf(sacc[nt][r] - mx);
      sacc[nt][r] = p;
      sum += p;
    }
    for (int off = 1; off < 16; off <<= 1) sum += __shfl_xor(sum, off, 16);
    float inv = 1.0f / sum;
#pragma unroll
    for (int nt = 0; nt < 5; ++nt) sacc[nt][r] *= inv;
  }
  // restage probs D-layout -> A-layout via LDS
#pragma unroll
  for (int nt = 0; nt < 5; ++nt)
#pragma unroll
    for (int r = 0; r < 8; ++r)
      Ps[wave][(r + 8 * hi) * 96 + nt * 16 + nl] = (__bf16)sacc[nt][r];
  __syncthreads();

  // O = P V : K=96 (3 chunks), N=48 (3 tiles)
  v8f oacc[3] = {};
#pragma unroll
  for (int kc = 0; kc < 3; ++kc) {
    v16bf pa = ldsLoadA(Ps[wave] + kc * 32, 96);
#pragma unroll
    for (int nt = 0; nt < 3; ++nt) {
      v16bf vb = ldsLoadB(Vt + nt * 16 * 96 + kc * 32, 96);
      oacc[nt] = WMMA_BF16(pa, vb, oacc[nt]);
    }
  }
#pragma unroll
  for (int nt = 0; nt < 3; ++nt) {
    int d = nt * 16 + nl;
    if (d < 40) {
#pragma unroll
      for (int r = 0; r < 8; ++r) {
        int s = s0 + wave * 16 + r + 8 * hi;
        attnbuf[(size_t)(b * 4096 + s) * 320 + h * 40 + d] = (__bf16)oacc[nt][r];
      }
    }
  }
}

// ---------------- Kernel 4: output projection + bias + residual ----------------
// out[b, c, s] = sum_n attn[b, s, n] * Wo[n, c] + bo[c] + hs[b, c, s]
__global__ __launch_bounds__(256) void k_oproj(const __bf16* __restrict__ attnbuf,
                                               const float* __restrict__ Wo,
                                               const float* __restrict__ bo,
                                               const float* __restrict__ hs,
                                               float* __restrict__ out) {
  __shared__ __bf16 As[128 * 32];
  __shared__ __bf16 Bs[64 * 32];
  const int s0 = blockIdx.x * 128;
  const int n0 = blockIdx.y * 64;
  const int b  = blockIdx.z;
  const int tid = threadIdx.x;
  const int wave = tid >> 5, lane = tid & 31;
  const int nl = lane & 15, hi = lane >> 4;

  v8f acc[4] = {};
  for (int kt = 0; kt < 10; ++kt) {
    const int k0 = kt * 32;
    // ---- A tile: pure bf16 2D tile copy (128 rows x 32 elems, row stride 320) ----
#if defined(__gfx1250__) && __has_builtin(__builtin_amdgcn_tensor_load_to_lds)
    if (wave == 0) {  // one TDM issue per workgroup (Tensor Data Mover, TENSORcnt)
      unsigned lds_addr = (unsigned)(uintptr_t)(&As[0]);  // LDS offset = low 32 bits of generic addr
      unsigned long long ga =
          (unsigned long long)(uintptr_t)(attnbuf + (size_t)(b * 4096 + s0) * 320 + k0);
      u32x4 g0 = {};
      g0[0] = 1u;                                   // count=1 user descriptor
      g0[1] = lds_addr;                             // D#.lds_addr
      g0[2] = (unsigned)(ga & 0xffffffffu);         // D#.global_addr[31:0]
      g0[3] = (unsigned)((ga >> 32) & 0x1ffffffu) | (2u << 30);  // addr[56:32] | type=2
      i32x8 g1 = {};
      g1[0] = (int)(1u << 16);                      // data_size=1 (2 bytes)
      g1[1] = (int)(320u << 16);                    // tensor_dim0 = 320
      g1[2] = (int)(4096u << 16);                   // tensor_dim1 = 4096
      g1[3] = (int)(32u << 16);                     // tile_dim0 = 32
      g1[4] = 128;                                  // tile_dim1 = 128
      g1[5] = 320;                                  // tensor_dim0_stride = 320
      g1[6] = 0;
      g1[7] = 0;
      i32x4 zg = {};
#if __clang_major__ >= 23
      i32x8 zg8 = {};
      __builtin_amdgcn_tensor_load_to_lds(g0, g1, zg, zg, zg8, 0);
#else
      __builtin_amdgcn_tensor_load_to_lds(g0, g1, zg, zg, 0);
#endif
      __builtin_amdgcn_s_wait_tensorcnt(0);
    }
#else
#pragma unroll
    for (int it = 0; it < 2; ++it) {   // fallback: v8bf (16B) chunk copies
      int idx = tid + it * 256;
      int c8 = idx & 3, si = idx >> 2;
      *(v8bf*)(As + si * 32 + c8 * 8) =
          *(const v8bf*)(attnbuf + (size_t)(b * 4096 + s0 + si) * 320 + k0 + c8 * 8);
    }
#endif
#pragma unroll
    for (int it = 0; it < 4; ++it) {   // B: 64 n x 16 kk-pairs
      int idx = tid + it * 256;
      int n = idx & 63, kk = (idx >> 6) * 2;
      st2bf(Bs + n * 32 + kk, Wo[(k0 + kk) * 320 + n0 + n], Wo[(k0 + kk + 1) * 320 + n0 + n]);
    }
    __syncthreads();
    v16bf a = ldsLoadA(As + wave * 16 * 32, 32);
#pragma unroll
    for (int nt = 0; nt < 4; ++nt) {
      v16bf bb = ldsLoadB(Bs + nt * 16 * 32, 32);
      acc[nt] = WMMA_BF16(a, bb, acc[nt]);
    }
    __syncthreads();
  }
  // epilogue: rows r are consecutive s -> float4 residual loads + float4 stores
#pragma unroll
  for (int nt = 0; nt < 4; ++nt) {
    int c = n0 + nt * 16 + nl;
    float bias = bo[c];
    size_t base = (size_t)(b * 320 + c) * 4096 + s0 + wave * 16 + 8 * hi;
    v4f h0 = *(const v4f*)(hs + base);
    v4f h1 = *(const v4f*)(hs + base + 4);
    v4f o0, o1;
#pragma unroll
    for (int r = 0; r < 4; ++r) {
      o0[r] = acc[nt][r]     + bias + h0[r];
      o1[r] = acc[nt][4 + r] + bias + h1[r];
    }
    *(v4f*)(out + base)     = o0;
    *(v4f*)(out + base + 4) = o1;
  }
}

extern "C" void kernel_launch(void* const* d_in, const int* in_sizes, int n_in,
                              void* d_out, int out_size, void* d_ws, size_t ws_size,
                              hipStream_t stream) {
  const float* hs  = (const float*)d_in[0];  // (16,320,64,64)
  const float* enc = (const float*)d_in[1];  // (16,93,768)
  const float* Wq  = (const float*)d_in[2];  // (320,320)
  const float* Wk  = (const float*)d_in[3];  // (768,320)
  const float* Wv  = (const float*)d_in[4];  // (768,320)
  const float* Wo  = (const float*)d_in[5];  // (320,320)
  const float* bo  = (const float*)d_in[6];  // (320,)
  float* out = (float*)d_out;

  char* ws = (char*)d_ws;
  const size_t Q_BYTES  = (size_t)16 * 4096 * 320 * 2;  // 41,943,040
  const size_t KV_BYTES = (size_t)16 * 80 * 320 * 2;    // 819,200
  __bf16* qbuf    = (__bf16*)(ws);
  __bf16* kbuf    = (__bf16*)(ws + Q_BYTES);
  __bf16* vbuf    = (__bf16*)(ws + Q_BYTES + KV_BYTES);
  __bf16* attnbuf = (__bf16*)(ws + Q_BYTES + 2 * KV_BYTES);

  k_qproj <<<dim3(32, 5, 16), 256, 0, stream>>>(hs, Wq, qbuf);
  k_kvproj<<<dim3(5, 2, 16), 160, 0, stream>>>(enc, Wk, Wv, kbuf, vbuf);
  k_attn  <<<dim3(64, 8, 16), 128, 0, stream>>>(qbuf, kbuf, vbuf, attnbuf);
  k_oproj <<<dim3(32, 5, 16), 256, 0, stream>>>(attnbuf, Wo, bo, hs, out);
}
